// MAB_12670153523686
// MI455X (gfx1250) — matrix operand order repaired and checked by reading, not compiled, not run on hardware
//
#include <hip/hip_runtime.h>
#include <hip/hip_bf16.h>

// ---------------------------------------------------------------------------
// MAB block for MI455X (gfx1250, wave32): WMMA 16x16x32 f16 (f32 accumulate)
// + Tensor Data Mover (TDM) async K/V tile staging in the attention kernel.
// ---------------------------------------------------------------------------

typedef __attribute__((ext_vector_type(16))) _Float16     v16h;
typedef __attribute__((ext_vector_type(8)))  float        v8f;
typedef __attribute__((ext_vector_type(4)))  float        v4f;
typedef __attribute__((ext_vector_type(4)))  unsigned int v4u;
typedef __attribute__((ext_vector_type(8)))  int          v8i;
typedef __attribute__((ext_vector_type(4)))  int          v4i;

union V16H { v16h v; v4u u[2]; _Float16 h[16]; };
union Pack8 { v4u u; _Float16 h[8]; };

#define WMMA_F16(A, B, C) \
    __builtin_amdgcn_wmma_f32_16x16x32_f16(false, (A), false, (B), (short)0, (C), false, false)

#if defined(__has_builtin)
#if __has_builtin(__builtin_amdgcn_tensor_load_to_lds) && __has_builtin(__builtin_amdgcn_s_wait_tensorcnt)
#define HAVE_TDM 1
#endif
#endif
#ifndef HAVE_TDM
#define HAVE_TDM 0
#endif

constexpr int NQ = 2048;   // query rows
constexpr int NK = 2048;   // key rows
constexpr int DV = 1024;   // model dim
constexpr int DK = 1024;   // input dim
constexpr int NH = 16;     // effective heads (DIM_SPLIT)
constexpr int HD = 64;     // head dim

// ---------------------------------------------------------------------------
// GEMM: C = epilogue( X[M,K] @ W[N,K]^T + bias[N] )
// ---------------------------------------------------------------------------
template <bool RELU, bool RES, bool SF32, bool SF16R, bool SF16T>
__global__ __launch_bounds__(128)
void gemm_proj(const float* __restrict__ X, const float* __restrict__ W,
               const float* __restrict__ bias, const float* __restrict__ resid,
               float* __restrict__ outF32, _Float16* __restrict__ outF16r,
               _Float16* __restrict__ outF16t, int M, int N, int K)
{
    __shared__ __align__(16) _Float16 Xs[64 * 40];
    __shared__ __align__(16) _Float16 Wsh[64 * 40];

    const int t    = threadIdx.x;
    const int wv   = t >> 5;
    const int lane = t & 31;
    const int hf   = lane >> 4;
    const int ln   = lane & 15;
    const int m0   = blockIdx.y * 64;
    const int n0   = blockIdx.x * 64;

    v8f acc[4] = {};

    for (int kt = 0; kt < K; kt += 32) {
        #pragma unroll
        for (int i = 0; i < 4; ++i) {
            int idx = t + 128 * i;
            int row = idx >> 3;
            int c0  = (idx & 7) * 4;
            v4f xv = *(const v4f*)&X[(size_t)(m0 + row) * K + kt + c0];
            v4f wq = *(const v4f*)&W[(size_t)(n0 + row) * K + kt + c0];
            _Float16* px = &Xs[row * 40 + c0];
            px[0] = (_Float16)xv.x; px[1] = (_Float16)xv.y;
            px[2] = (_Float16)xv.z; px[3] = (_Float16)xv.w;
            _Float16* pw = &Wsh[row * 40 + c0];
            pw[0] = (_Float16)wq.x; pw[1] = (_Float16)wq.y;
            pw[2] = (_Float16)wq.z; pw[3] = (_Float16)wq.w;
        }
        __syncthreads();

        V16H a;
        const _Float16* ar = &Xs[(16 * wv + ln) * 40 + 8 * hf];
        a.u[0] = *(const v4u*)ar;
        a.u[1] = *(const v4u*)(ar + 16);

        #pragma unroll
        for (int nb = 0; nb < 4; ++nb) {
            const _Float16* br = &Wsh[(16 * nb + ln) * 40 + 8 * hf];
            V16H b;
            b.u[0] = *(const v4u*)br;
            b.u[1] = *(const v4u*)(br + 16);
            acc[nb] = WMMA_F16(a.v, b.v, acc[nb]);
        }
        __syncthreads();
    }

    #pragma unroll
    for (int nb = 0; nb < 4; ++nb) {
        int col  = n0 + 16 * nb + ln;
        float bv = bias[col];
        float vals[8];
        #pragma unroll
        for (int j = 0; j < 8; ++j) {
            int row = m0 + 16 * wv + 8 * hf + j;
            float o = acc[nb][j] + bv;
            if (RELU) o = fmaxf(o, 0.0f);
            if (RES)  o += resid[(size_t)row * N + col];
            vals[j] = o;
            if (SF32)  outF32[(size_t)row * N + col]  = o;
            if (SF16R) outF16r[(size_t)row * N + col] = (_Float16)o;
        }
        if (SF16T) {
            Pack8 pk;
            #pragma unroll
            for (int j = 0; j < 8; ++j) pk.h[j] = (_Float16)vals[j];
            int rowb = m0 + 16 * wv + 8 * hf;
            *(v4u*)&outF16t[(size_t)col * M + rowb] = pk.u;
        }
    }
}

// ---------------------------------------------------------------------------
// TDM: issue an async 2-D tile load (f16 elements) global -> LDS.
// D# per CDNA5 ISA ch.8: group0 = {count, lds_addr, global_addr, type=2},
// group1 = {data_size=2B, tensor dims/strides, tile dims}, groups 2/3 and
// the extra group zero (tile_dim2 = 0 => 2-D tensor). Tracked by TENSORcnt.
// This toolchain exposes the 6-arg builtin:
//   (uint32x4 g0, int32x8 g1, int32x4 g2, int32x4 g3, int32x8 extra, i32 cpol)
// ---------------------------------------------------------------------------
#if HAVE_TDM
__device__ __forceinline__ void tdm_load_2d_f16(unsigned lds_off, const _Float16* gptr,
                                                unsigned tdim0, unsigned tdim1,
                                                unsigned stride0,
                                                unsigned tile0, unsigned tile1)
{
    unsigned long long ga = (unsigned long long)(size_t)gptr;
    v4u g0;
    g0[0] = 1u;                                              // count = 1 valid D#
    g0[1] = lds_off;                                         // LDS byte address
    g0[2] = (unsigned)(ga & 0xFFFFFFFFu);                    // global_addr[31:0]
    g0[3] = (unsigned)((ga >> 32) & 0x01FFFFFFu) | (2u << 30); // [56:32] | type=2
    v8i g1;
    g1[0] = (int)(1u << 16);                                 // data_size=1 (2 bytes)
    g1[1] = (int)((tdim0 & 0xFFFFu) << 16);                  // tensor_dim0[15:0]
    g1[2] = (int)((tdim0 >> 16) | ((tdim1 & 0xFFFFu) << 16));// dim0[31:16]|dim1[15:0]
    g1[3] = (int)((tdim1 >> 16) | (tile0 << 16));            // dim1[31:16]|tile_dim0
    g1[4] = (int)(tile1 & 0xFFFFu);                          // tile_dim1, tile_dim2=0
    g1[5] = (int)stride0;                                    // tensor_dim0_stride
    g1[6] = 0;
    g1[7] = 0;
    v4i z4 = {0, 0, 0, 0};
    v8i z8 = {0, 0, 0, 0, 0, 0, 0, 0};
    __builtin_amdgcn_tensor_load_to_lds(g0, g1, z4, z4, z8, 0);
}
#endif

__device__ __forceinline__ unsigned lds_offset_of(const void* p)
{
    // flat LDS address: addr[31:0] is the LDS byte offset (ISA 10.2 aperture calc)
    return (unsigned)(size_t)p;
}

// ---------------------------------------------------------------------------
// Flash attention, 4 waves / block, one head x 64 q-rows per block.
// K/V tiles (32 keys) staged in LDS, double-buffered via TDM async copies.
// ---------------------------------------------------------------------------
__global__ __launch_bounds__(128)
void attn_flash(const _Float16* __restrict__ q16, const _Float16* __restrict__ k16,
                const _Float16* __restrict__ vtr, const float* __restrict__ q32,
                float* __restrict__ O)
{
    __shared__ __align__(16) _Float16 Ks[2][32 * 64];   // [key][d]  per buffer
    __shared__ __align__(16) _Float16 Vs[2][64 * 32];   // [dcol][key] per buffer
    __shared__ __align__(16) _Float16 Pl[4][16 * 32];   // per-wave P bounce

    const int t    = threadIdx.x;
    const int wv   = t >> 5;
    const int lane = t & 31;
    const int hf   = lane >> 4;
    const int ln   = lane & 15;
    const int h    = blockIdx.x >> 5;                 // 32 q-blocks per head
    const int m0   = (blockIdx.x & 31) * 64 + 16 * wv;

    // Q A-operands for this wave's 16 rows (two K=32 chunks over d=0..63)
    V16H aq[2];
    #pragma unroll
    for (int c = 0; c < 2; ++c) {
        const _Float16* p = &q16[(size_t)(m0 + ln) * DV + h * HD + 32 * c + 8 * hf];
        aq[c].u[0] = *(const v4u*)p;
        aq[c].u[1] = *(const v4u*)(p + 16);
    }

    float mst[8], lst[8];
    #pragma unroll
    for (int j = 0; j < 8; ++j) { mst[j] = -1.0e30f; lst[j] = 0.0f; }
    v8f oacc[4] = {};

#if HAVE_TDM
    if (wv == 0) {   // prologue: stage tiles for kk = 0 into buffer 0
        tdm_load_2d_f16(lds_offset_of(&Ks[0][0]),
                        k16 + (size_t)h * HD, DV, NK, DV, HD, 32);
        tdm_load_2d_f16(lds_offset_of(&Vs[0][0]),
                        vtr + (size_t)(h * HD) * NK, NK, DV, NK, 32, HD);
    }
#endif

    for (int kk = 0; kk < NK; kk += 32) {
        const int buf = (kk >> 5) & 1;

#if HAVE_TDM
        if (wv == 0) {
            if (kk + 32 < NK) {   // prefetch next tiles into the other buffer
                tdm_load_2d_f16(lds_offset_of(&Ks[buf ^ 1][0]),
                                k16 + (size_t)(kk + 32) * DV + h * HD,
                                DV, NK, DV, HD, 32);
                tdm_load_2d_f16(lds_offset_of(&Vs[buf ^ 1][0]),
                                vtr + (size_t)(h * HD) * NK + (kk + 32),
                                NK, DV, NK, 32, HD);
                __builtin_amdgcn_s_wait_tensorcnt(2);  // in-order: current 2 landed
            } else {
                __builtin_amdgcn_s_wait_tensorcnt(0);
            }
        }
        __syncthreads();   // tiles in Ks/Vs[buf] visible to all 4 waves
#else
        // fallback: synchronous cooperative staging (single buffer semantics)
        __syncthreads();
        {
            const v4u* gk = (const v4u*)(k16 + (size_t)kk * DV + h * HD);
            const v4u* gv = (const v4u*)(vtr + (size_t)(h * HD) * NK + kk);
            #pragma unroll
            for (int i = 0; i < 2; ++i) {
                int idx = t + 128 * i;             // 0..255 chunks of 8 halves
                int kr  = idx >> 3, kc = idx & 7;  // K tile: 32 rows x 64 halves
                ((v4u*)&Ks[buf][kr * 64])[kc] = gk[(size_t)kr * (DV / 8) + kc];
                int vr = idx >> 2, vc = idx & 3;   // V tile: 64 rows x 32 halves
                ((v4u*)&Vs[buf][vr * 32])[vc] = gv[(size_t)vr * (NK / 8) + vc];
            }
        }
        __syncthreads();
#endif

        // ---- S tiles: 16 q-rows x 32 keys (two 16x16 f32 accumulators) ----
        v8f s[2];
        #pragma unroll
        for (int ks = 0; ks < 2; ++ks) {
            v8f c = {};
            #pragma unroll
            for (int cc = 0; cc < 2; ++cc) {
                const _Float16* p = &Ks[buf][(16 * ks + ln) * 64 + 32 * cc + 8 * hf];
                V16H b;
                b.u[0] = *(const v4u*)p;
                b.u[1] = *(const v4u*)(p + 16);
                c = WMMA_F16(aq[cc].v, b.v, c);
            }
            s[ks] = c;
        }

        // ---- online softmax update ----
        float mnew[8], sc[8];
        #pragma unroll
        for (int j = 0; j < 8; ++j) {
            float tmax = fmaxf(s[0][j], s[1][j]);
            tmax = fmaxf(tmax, __shfl_xor(tmax, 1));
            tmax = fmaxf(tmax, __shfl_xor(tmax, 2));
            tmax = fmaxf(tmax, __shfl_xor(tmax, 4));
            tmax = fmaxf(tmax, __shfl_xor(tmax, 8));
            mnew[j] = fmaxf(mst[j], tmax);
            sc[j]   = __expf(mst[j] - mnew[j]);
        }
        v8f p0, p1;
        #pragma unroll
        for (int j = 0; j < 8; ++j) {
            p0[j] = __expf(s[0][j] - mnew[j]);
            p1[j] = __expf(s[1][j] - mnew[j]);
            float r = p0[j] + p1[j];
            r += __shfl_xor(r, 1);
            r += __shfl_xor(r, 2);
            r += __shfl_xor(r, 4);
            r += __shfl_xor(r, 8);
            lst[j] = lst[j] * sc[j] + r;
            mst[j] = mnew[j];
        }
        #pragma unroll
        for (int dc = 0; dc < 4; ++dc)
            #pragma unroll
            for (int j = 0; j < 8; ++j) oacc[dc][j] *= sc[j];

        // ---- relayout P (C layout) -> A layout via per-wave LDS bounce ----
        #pragma unroll
        for (int j = 0; j < 8; ++j) {
            Pl[wv][(j + 8 * hf) * 32 + ln]      = (_Float16)p0[j];
            Pl[wv][(j + 8 * hf) * 32 + ln + 16] = (_Float16)p1[j];
        }
        __syncthreads();
        V16H ap;
        const _Float16* pp = &Pl[wv][ln * 32 + 8 * hf];
        ap.u[0] = *(const v4u*)pp;
        ap.u[1] = *(const v4u*)(pp + 16);

        // ---- O(16x64) += P(16x32) @ V(32x64) from LDS V tile ----
        #pragma unroll
        for (int dc = 0; dc < 4; ++dc) {
            const _Float16* pv = &Vs[buf][(16 * dc + ln) * 32 + 8 * hf];
            V16H b;
            b.u[0] = *(const v4u*)pv;
            b.u[1] = *(const v4u*)(pv + 16);
            oacc[dc] = WMMA_F16(ap.v, b.v, oacc[dc]);
        }
        __syncthreads();   // all reads of buf done before next TDM overwrite
    }

    // ---- finalize: O = q_f32 + acc / (l * sqrt(DV)) ----
    float inv[8];
    #pragma unroll
    for (int j = 0; j < 8; ++j) inv[j] = 1.0f / (lst[j] * 32.0f);
    #pragma unroll
    for (int dc = 0; dc < 4; ++dc) {
        int col = h * HD + 16 * dc + ln;
        #pragma unroll
        for (int j = 0; j < 8; ++j) {
            int row = m0 + 8 * hf + j;
            O[(size_t)row * DV + col] = q32[(size_t)row * DV + col] + oacc[dc][j] * inv[j];
        }
    }
}

// ---------------------------------------------------------------------------
extern "C" void kernel_launch(void* const* d_in, const int* in_sizes, int n_in,
                              void* d_out, int out_size, void* d_ws, size_t ws_size,
                              hipStream_t stream)
{
    (void)in_sizes; (void)n_in; (void)out_size; (void)ws_size;

    const float* Q  = (const float*)d_in[0];
    const float* Kx = (const float*)d_in[1];
    const float* Wq = (const float*)d_in[2];
    const float* bq = (const float*)d_in[3];
    const float* Wk = (const float*)d_in[4];
    const float* bk = (const float*)d_in[5];
    const float* Wv = (const float*)d_in[6];
    const float* bv = (const float*)d_in[7];
    const float* Wo = (const float*)d_in[8];
    const float* bo = (const float*)d_in[9];
    float* out = (float*)d_out;

    char* w = (char*)d_ws;
    float*    q32 = (float*)w;    w += (size_t)NQ * DV * 4;
    _Float16* q16 = (_Float16*)w; w += (size_t)NQ * DV * 2;
    _Float16* k16 = (_Float16*)w; w += (size_t)NK * DV * 2;
    _Float16* vtr = (_Float16*)w; w += (size_t)DV * NK * 2;   // [DV][NK]
    float*    Oat = (float*)w;    w += (size_t)NQ * DV * 4;

    dim3 blk(128);
    dim3 gq(DV / 64, NQ / 64);
    dim3 gk(DV / 64, NK / 64);

    gemm_proj<false, false, true, true, false><<<gq, blk, 0, stream>>>(
        Q, Wq, bq, (const float*)nullptr, q32, q16, (_Float16*)nullptr, NQ, DV, DK);
    gemm_proj<false, false, false, true, false><<<gk, blk, 0, stream>>>(
        Kx, Wk, bk, (const float*)nullptr, (float*)nullptr, k16, (_Float16*)nullptr, NK, DV, DK);
    gemm_proj<false, false, false, false, true><<<gk, blk, 0, stream>>>(
        Kx, Wv, bv, (const float*)nullptr, (float*)nullptr, (_Float16*)nullptr, vtr, NK, DV, DK);

    attn_flash<<<NH * (NQ / 64), 128, 0, stream>>>(q16, k16, vtr, q32, Oat);

    gemm_proj<true, true, true, false, false><<<gq, blk, 0, stream>>>(
        Oat, Wo, bo, Oat, out, (_Float16*)nullptr, (_Float16*)nullptr, NQ, DV, DV);
}
